// AttentionGRUCell_59141699666631
// MI455X (gfx1250) — matrix operand order repaired and checked by reading, not compiled, not run on hardware
//
#include <hip/hip_runtime.h>

// AttentionGRU scan for MI455X (gfx1250, wave32, WMMA).
// softmax over singleton axis == 1  =>  z_hat == x_t, attention params dead.
// Per step (per 16-row batch slab):
//   gx = x_t @ (kernel + attention_kernel) + bias
//   rh = h @ Rk[:, :256]
//   z = sigmoid(gx+rh)[:, :128]; r = sigmoid(gx+rh)[:, 128:256]
//   hh = tanh(gx[:, 256:] + (r*h) @ Rk[:, 256:])
//   h  = z*h + (1-z)*hh
// 8 WGs x 16 batch rows. ALL matrix operands held in LDS pre-packed into the
// exact WMMA per-lane register layout => every fragment = 2x ds_load_b128.

#define B_  128
#define T_  2000
#define D_  64
#define U_  128
#define NU3 384

typedef __attribute__((ext_vector_type(16))) __bf16        v16bf;
typedef __attribute__((ext_vector_type(8)))  float         v8f;
typedef __attribute__((ext_vector_type(4)))  unsigned int  v4u;

union AF { v16bf v; unsigned short u[16]; v4u q[2]; };

__device__ __forceinline__ unsigned short f2bf(float f) {
  unsigned int x = __builtin_bit_cast(unsigned int, f);
  x += 0x7FFFu + ((x >> 16) & 1u);     // round-to-nearest-even
  return (unsigned short)(x >> 16);
}

__device__ __forceinline__ float sigm(float x) { return 1.0f / (1.0f + __expf(-x)); }

// Packed fragment storage: frag = 512 ushorts = 64 slots of 16B.
// Lane L reads slot L (half 0) and slot 32+L (half 1): two ds_load_b128,
// 16B stride across lanes -> conflict-free.
__device__ __forceinline__ AF loadFrag(const v4u* P, int frag, int lane) {
  AF a;
  a.q[0] = P[frag * 64 + lane];
  a.q[1] = P[frag * 64 + 32 + lane];
  return a;
}

// ushort index of element e (0..15) of lane within a packed fragment
__device__ __forceinline__ int fragIdx(int frag, int lane, int e) {
  return frag * 512 + (e >> 3) * 256 + lane * 8 + (e & 7);
}

__device__ __forceinline__ v8f wmma_bf16(AF a, AF b, v8f c) {
  return __builtin_amdgcn_wmma_f32_16x16x32_bf16(false, a.v, false, b.v,
                                                 (short)0, c, false, false);
}

__global__ __launch_bounds__(384)
void attgru_scan(const float* __restrict__ x,
                 const float* __restrict__ kernel,
                 const float* __restrict__ rkernel,
                 const float* __restrict__ akernel,
                 const float* __restrict__ bias,
                 const float* __restrict__ h0,
                 float* __restrict__ out) {
  // ~173.5 KB LDS (CDNA5: 320 KB/WGP)
  __shared__ v4u  sWcP[48 * 64];   // Wc B-frags: 24 tiles x 2 K-chunks, 48 KB
  __shared__ v4u  sRkP[96 * 64];   // Rk B-frags: 24 tiles x 4 K-chunks, 96 KB
  __shared__ v4u  sHP [4 * 64];    // h   A-frags (4 K-chunks), bf16 packed
  __shared__ v4u  sRPP[4 * 64];    // r*h A-frags (4 K-chunks), bf16 packed
  __shared__ v4u  sXP [2][2 * 64]; // x_t A-frags (2 K-chunks), double-buffered
  __shared__ float sH[16 * U_];    // h slab f32 (for r*h product and blend)
  __shared__ float sZ[16 * U_];    // update gate
  __shared__ float sBias[NU3];

  const int tid  = threadIdx.x;
  const int lane = tid & 31;
  const int wid  = tid >> 5;       // 12 waves
  const int row  = lane & 15;
  const int hi   = lane >> 4;
  const int b0   = blockIdx.x * 16;

  // ---- one-time init: pack weights straight into WMMA fragment layout ----
  // B-frag element e of lane:  K = 32*kc + 16*(e>>3) + 8*(lane>=16) + (e&7),  col = 16*tile + (lane&15)
  {
    unsigned short* wcp = (unsigned short*)sWcP;
    for (int i = tid; i < 48 * 512; i += 384) {
      int frag = i >> 9, rem = i & 511;
      int half = rem >> 8, ln = (rem >> 3) & 31, e7 = rem & 7;
      int tile = frag >> 1, kc = frag & 1;
      int K   = 32 * kc + 16 * half + 8 * (ln >> 4) + e7;
      int col = tile * 16 + (ln & 15);
      wcp[i] = f2bf(kernel[K * NU3 + col] + akernel[K * NU3 + col]);
    }
    unsigned short* rkp = (unsigned short*)sRkP;
    for (int i = tid; i < 96 * 512; i += 384) {
      int frag = i >> 9, rem = i & 511;
      int half = rem >> 8, ln = (rem >> 3) & 31, e7 = rem & 7;
      int tile = frag >> 2, kc = frag & 3;
      int K   = 32 * kc + 16 * half + 8 * (ln >> 4) + e7;
      int col = tile * 16 + (ln & 15);
      rkp[i] = f2bf(rkernel[K * NU3 + col]);
    }
    // A-frag element e of lane:  K = 32*kc + 16*(e>>3) + 8*(lane>=16) + (e&7), row = lane&15
    unsigned short* hp = (unsigned short*)sHP;
    for (int i = tid; i < 4 * 512; i += 384) {
      int kc = i >> 9, rem = i & 511;
      int half = rem >> 8, ln = (rem >> 3) & 31, e7 = rem & 7;
      int K = 32 * kc + 16 * half + 8 * (ln >> 4) + e7;
      hp[i] = f2bf(h0[(size_t)(b0 + (ln & 15)) * U_ + K]);
    }
    unsigned short* xp = (unsigned short*)sXP[0];
    for (int i = tid; i < 2 * 512; i += 384) {
      int kc = i >> 9, rem = i & 511;
      int half = rem >> 8, ln = (rem >> 3) & 31, e7 = rem & 7;
      int K = 32 * kc + 16 * half + 8 * (ln >> 4) + e7;
      xp[i] = f2bf(x[(size_t)(b0 + (ln & 15)) * (size_t)T_ * D_ + K]);
    }
    for (int i = tid; i < NU3;     i += 384) sBias[i] = bias[i];
    for (int i = tid; i < 16 * U_; i += 384)
      sH[i] = h0[(size_t)(b0 + (i >> 7)) * U_ + (i & 127)];
  }
  __syncthreads();

  int buf = 0;
  for (int t = 0; t < T_; ++t) {
    // ================= phase A: gx (+ rh for gate tiles) =================
    AF ah[4];
    if (wid < 8) {
#pragma unroll
      for (int kc = 0; kc < 4; ++kc) ah[kc] = loadFrag(sHP, kc, lane);
    }
    AF ax[2];
#pragma unroll
    for (int kc = 0; kc < 2; ++kc) ax[kc] = loadFrag(sXP[buf], kc, lane);

    v8f acc0 = {}, acc1 = {};
#pragma unroll
    for (int s = 0; s < 2; ++s) {
      const int tile = 2 * wid + s;          // wave-uniform
      v8f acc = {};
#pragma unroll
      for (int kc = 0; kc < 2; ++kc)         // x @ Wc  (K = 64)
        acc = wmma_bf16(ax[kc], loadFrag(sWcP, tile * 2 + kc, lane), acc);
      if (tile < 16) {                       // + h @ Rk[:, :256]  (K = 128)
#pragma unroll
        for (int kc = 0; kc < 4; ++kc)
          acc = wmma_bf16(ah[kc], loadFrag(sRkP, tile * 4 + kc, lane), acc);
      }
      const float bb = sBias[tile * 16 + row];
#pragma unroll
      for (int v = 0; v < 8; ++v) acc[v] += bb;
      if (s == 0) acc0 = acc; else acc1 = acc;
    }

    if (wid < 4) {                           // z gate, cols 0..127
#pragma unroll
      for (int s = 0; s < 2; ++s) {
        v8f acc = s ? acc1 : acc0;
        const int col = (2 * wid + s) * 16 + row;
#pragma unroll
        for (int v = 0; v < 8; ++v)
          sZ[(v + 8 * hi) * U_ + col] = sigm(acc[v]);
      }
    } else if (wid < 8) {                    // r gate -> scatter r*h into packed A-frags
      unsigned short* rpp = (unsigned short*)sRPP;
#pragma unroll
      for (int s = 0; s < 2; ++s) {
        v8f acc = s ? acc1 : acc0;
        const int c   = (2 * wid + s) * 16 - 128 + row;  // h column (K index)
        const int kcA = c >> 5, cc = c & 31;
        const int e   = (cc >> 4) * 8 + (cc & 7);        // half*8 + e7
        const int hiA = (cc >> 3) & 1;
#pragma unroll
        for (int v = 0; v < 8; ++v) {
          const int m = v + 8 * hi;
          rpp[fragIdx(kcA, m + 16 * hiA, e)] =
              f2bf(sigm(acc[v]) * sH[m * U_ + c]);
        }
      }
    }
    __syncthreads();

    // ================= phase B =================
    if (wid >= 8) {                          // finish hh, blend, update h, store
      const int j = wid - 8;
      AF ar[4];
#pragma unroll
      for (int kc = 0; kc < 4; ++kc) ar[kc] = loadFrag(sRPP, kc, lane);
      unsigned short* hp = (unsigned short*)sHP;
#pragma unroll
      for (int s = 0; s < 2; ++s) {
        const int n0h = 32 * j + 16 * s;     // hh column tile
        v8f acc = s ? acc1 : acc0;           // carried gx[:, 256+n0h ...]
#pragma unroll
        for (int kc = 0; kc < 4; ++kc)       // + (r*h) @ Rk[:, 256:]
          acc = wmma_bf16(ar[kc],
                          loadFrag(sRkP, (16 + (n0h >> 4)) * 4 + kc, lane), acc);
        const int c   = n0h + row;
        const int kcA = c >> 5, cc = c & 31;
        const int e   = (cc >> 4) * 8 + (cc & 7);
        const int hiA = (cc >> 3) & 1;
#pragma unroll
        for (int v = 0; v < 8; ++v) {
          const int m  = v + 8 * hi;
          const float hh = tanhf(acc[v]);
          const float z  = sZ[m * U_ + c];
          const float hn = z * sH[m * U_ + c] + (1.0f - z) * hh;
          sH[m * U_ + c] = hn;
          hp[fragIdx(kcA, m + 16 * hiA, e)] = f2bf(hn);
          out[(size_t)(b0 + m) * (size_t)T_ * U_ + (size_t)t * U_ + c] = hn;
        }
      }
    } else {                                 // waves 0..7: stage x_{t+1} packed, prefetch t+2
      if (t + 1 < T_) {
        unsigned short* xp = (unsigned short*)sXP[buf ^ 1];
        for (int i = tid; i < 1024; i += 256) {
          int kc = i >> 9, rem = i & 511;
          int half = rem >> 8, ln = (rem >> 3) & 31, e7 = rem & 7;
          int K = 32 * kc + 16 * half + 8 * (ln >> 4) + e7;
          xp[i] = f2bf(x[(size_t)(b0 + (ln & 15)) * (size_t)T_ * D_ +
                         (size_t)(t + 1) * D_ + K]);
        }
      }
      if (t + 2 < T_ && tid < 16) {
        __builtin_prefetch(&x[(size_t)(b0 + tid) * (size_t)T_ * D_ +
                              (size_t)(t + 2) * D_], 0, 1);
      }
    }
    __syncthreads();
    buf ^= 1;
  }
}

extern "C" void kernel_launch(void* const* d_in, const int* in_sizes, int n_in,
                              void* d_out, int out_size, void* d_ws, size_t ws_size,
                              hipStream_t stream) {
  (void)in_sizes; (void)n_in; (void)d_ws; (void)ws_size; (void)out_size;
  const float* x       = (const float*)d_in[0];
  const float* kernel  = (const float*)d_in[1];
  const float* rkernel = (const float*)d_in[2];
  const float* akernel = (const float*)d_in[3];
  // d_in[4..5], d_in[7..8]: attention_w/u/b/v — numerically inert (alpha == 1)
  const float* bias    = (const float*)d_in[6];
  const float* h0      = (const float*)d_in[9];
  float* out           = (float*)d_out;

  attgru_scan<<<B_ / 16, 384, 0, stream>>>(x, kernel, rkernel, akernel, bias, h0, out);
}
